// SpectralConv1d_12317966205470
// MI455X (gfx1250) — matrix operand order repaired and checked by reading, not compiled, not run on hardware
//
#include <hip/hip_runtime.h>
#include <hip/hip_bf16.h>
#include <math.h>

typedef __bf16 bf16_t;
typedef __attribute__((ext_vector_type(16))) __bf16 v16bf;
typedef __attribute__((ext_vector_type(8)))  __bf16 v8bf;
typedef __attribute__((ext_vector_type(8)))  float  v8f;

#define BATCH 65536
#define CH    1024
#define MODES 128

#define TWO_PI 6.2831853071795864769f

// ------------------------------------------------------------------
// Kernel 1: fold complex weights with the irfft basis.
//   Gr[n,c'] =  s_n*cos(2pi n c'/1024)/1024,  Gi[n,c'] = -s_n*sin(...)/1024
//   (s_0 = 1, s_n = 2 for n >= 1; modes < 512 so no Nyquist term)
//   P[m,c'] = sum_n wr*Gr + wi*Gi      Q[m,c'] = sum_n wr*Gi - wi*Gr
// ------------------------------------------------------------------
__global__ __launch_bounds__(256) void build_pq(const float* __restrict__ w,
                                                float* __restrict__ P,
                                                float* __restrict__ Q) {
  __shared__ float tc[1024], ts[1024];
  const int tid = threadIdx.x;
  for (int i = tid; i < 1024; i += 256) {
    float s, c;
    __sincosf(TWO_PI * (float)i * (1.0f / 1024.0f), &s, &c);
    tc[i] = c; ts[i] = s;
  }
  __syncthreads();
  const int m  = blockIdx.x;                 // mode (uniform per block -> scalar loads)
  const int cp = blockIdx.y * 256 + tid;     // output channel
  float p = 0.0f, q = 0.0f;
  for (int n = 0; n < MODES; ++n) {
    const float sc = (n == 0) ? (1.0f / 1024.0f) : (2.0f / 1024.0f);
    const float wr = w[(m * MODES + n) * 2 + 0] * sc;
    const float wi = w[(m * MODES + n) * 2 + 1] * sc;
    const int idx = (n * cp) & 1023;
    const float gc = tc[idx], gs = ts[idx];
    p += wr * gc - wi * gs;                  // wr*Gr + wi*Gi
    q += -wr * gs - wi * gc;                 // wr*Gi - wi*Gr
  }
  P[m * 1024 + cp] = p;
  Q[m * 1024 + cp] = q;
}

// ------------------------------------------------------------------
// Kernel 2: fuse the rfft basis -> real 1024x1024 matrix, stored
// transposed (Mt[c'][c]) in bf16 so B-fragments are contiguous in K.
//   M[c,c'] = sum_m cos(2pi m c/1024)*P[m,c'] - sin(...)*Q[m,c']
// ------------------------------------------------------------------
__global__ __launch_bounds__(256) void build_m(const float* __restrict__ P,
                                               const float* __restrict__ Q,
                                               bf16_t* __restrict__ Mt) {
  __shared__ float tc[1024], ts[1024];
  const int tid = threadIdx.x;
  for (int i = tid; i < 1024; i += 256) {
    float s, c;
    __sincosf(TWO_PI * (float)i * (1.0f / 1024.0f), &s, &c);
    tc[i] = c; ts[i] = s;
  }
  __syncthreads();
  const int n = blockIdx.x;                  // output channel (uniform -> scalar P/Q loads)
  const int c = blockIdx.y * 256 + tid;      // input channel
  float acc = 0.0f;
  for (int m = 0; m < MODES; ++m) {
    const float p = P[m * 1024 + n];
    const float q = Q[m * 1024 + n];
    const int idx = (m * c) & 1023;
    acc += tc[idx] * p - ts[idx] * q;
  }
  Mt[(size_t)n * 1024 + c] = (bf16_t)acc;
}

// ------------------------------------------------------------------
// gfx1250 async global->LDS copy (ASYNCcnt tracked), 16 bytes per lane.
// ------------------------------------------------------------------
__device__ __forceinline__ void async_b128(const void* g, void* l) {
  asm volatile("global_load_async_to_lds_b128 %0, %1, off"
               :: "v"((__attribute__((address_space(3))) void*)l),
                  "v"(g)
               : "memory");
}
__device__ __forceinline__ void wait_async0() {
  asm volatile("s_wait_asynccnt 0x0" ::: "memory");
}

// ------------------------------------------------------------------
// Kernel 3: out = x @ M via bf16 WMMA. Block tile 128x128, K-step 32.
// 8 waves/block; wave tile 32x64 (2x4 of 16x16) -> 8 WMMA per K-step.
// Double-buffered LDS (one barrier per step):
//   - Mt tile:  global_load_async_to_lds_b128 (direct L2->LDS, ASYNCcnt)
//   - x  tile:  global f32 loads -> v_cvt_pk_bf16_f32 -> ds_store
// LDS rows padded to 40 halfs (80 B): 16-B aligned, conflict-free b128.
// ------------------------------------------------------------------
#define BM 128
#define BN 128
#define BK 32
#define LSTR 40
#define XTILE (BM * LSTR)
#define MTILE (BN * LSTR)

__global__ __launch_bounds__(256) void spectral_gemm(const float* __restrict__ x,
                                                     const bf16_t* __restrict__ Mt,
                                                     float* __restrict__ out) {
  __shared__ __align__(16) bf16_t xs[2 * XTILE];
  __shared__ __align__(16) bf16_t ms[2 * MTILE];

  const int tid  = threadIdx.x;
  const int lane = tid & 31;
  const int wid  = tid >> 5;
  const int b0 = blockIdx.y * BM;
  const int n0 = blockIdx.x * BN;
  const int rowg = wid & 3;     // 4 row groups x 32 rows
  const int colg = wid >> 2;    // 2 col groups x 64 cols
  const int lh = lane & 15;
  const int lg = lane >> 4;

  v8f acc[2][4];
#pragma unroll
  for (int i = 0; i < 2; ++i)
#pragma unroll
    for (int j = 0; j < 4; ++j)
      acc[i][j] = (v8f){0.f, 0.f, 0.f, 0.f, 0.f, 0.f, 0.f, 0.f};

  const int sr = tid >> 1;         // staging row 0..127 (2 threads per row)
  const int sh = (tid & 1) * 16;   // staging element offset 0/16
  const float*  xsrc = x  + (size_t)(b0 + sr) * CH + sh;
  const bf16_t* msrc = Mt + (size_t)(n0 + sr) * CH + sh;
  bf16_t* xdst = xs + sr * LSTR + sh;
  bf16_t* mdst = ms + sr * LSTR + sh;

  float4 fx[4];
  auto load_x = [&](int k0) {
    const float4* src = (const float4*)(xsrc + k0);
    fx[0] = src[0]; fx[1] = src[1]; fx[2] = src[2]; fx[3] = src[3];
  };
  auto store_x = [&](int buf) {
    v8bf lo, hi;
    lo[0] = (bf16_t)fx[0].x; lo[1] = (bf16_t)fx[0].y; lo[2] = (bf16_t)fx[0].z; lo[3] = (bf16_t)fx[0].w;
    lo[4] = (bf16_t)fx[1].x; lo[5] = (bf16_t)fx[1].y; lo[6] = (bf16_t)fx[1].z; lo[7] = (bf16_t)fx[1].w;
    hi[0] = (bf16_t)fx[2].x; hi[1] = (bf16_t)fx[2].y; hi[2] = (bf16_t)fx[2].z; hi[3] = (bf16_t)fx[2].w;
    hi[4] = (bf16_t)fx[3].x; hi[5] = (bf16_t)fx[3].y; hi[6] = (bf16_t)fx[3].z; hi[7] = (bf16_t)fx[3].w;
    *(v8bf*)(xdst + buf * XTILE)     = lo;
    *(v8bf*)(xdst + buf * XTILE + 8) = hi;
  };
  auto issue_m = [&](int k0, int buf) {
    async_b128(msrc + k0,     mdst + buf * MTILE);
    async_b128(msrc + k0 + 8, mdst + buf * MTILE + 8);
  };

  // ---- prologue: stage tile 0 into buffer 0 ----
  load_x(0);
  issue_m(0, 0);
  store_x(0);
  wait_async0();
  __syncthreads();

  int buf = 0;
  for (int k0 = 0; k0 < CH; k0 += BK) {
    const int nb = buf ^ 1;
    const bool more = (k0 + BK) < CH;
    if (more) {
      load_x(k0 + BK);                 // global f32 loads for next tile (hidden by WMMA)
      issue_m(k0 + BK, nb);            // async L2 -> LDS for next Mt tile
      if (k0 + 2 * BK < CH)
        __builtin_prefetch(xsrc + k0 + 2 * BK, 0, 1);
    }

    // ---- preload ALL fragments of the current buffer, then 8 WMMAs ----
    const bf16_t* xb = xs + buf * XTILE;
    const bf16_t* mb = ms + buf * MTILE;
    v16bf afrag[2], bfrag[4];
#pragma unroll
    for (int rt = 0; rt < 2; ++rt) {
      const int rr = rowg * 32 + rt * 16 + lh;
      v8bf alo = *(const v8bf*)(xb + rr * LSTR + lg * 8);
      v8bf ahi = *(const v8bf*)(xb + rr * LSTR + lg * 8 + 16);
      afrag[rt] = __builtin_shufflevector(alo, ahi,
                    0, 1, 2, 3, 4, 5, 6, 7, 8, 9, 10, 11, 12, 13, 14, 15);
    }
#pragma unroll
    for (int ct = 0; ct < 4; ++ct) {
      const int nn = colg * 64 + ct * 16 + lh;
      v8bf blo = *(const v8bf*)(mb + nn * LSTR + lg * 8);
      v8bf bhi = *(const v8bf*)(mb + nn * LSTR + lg * 8 + 16);
      bfrag[ct] = __builtin_shufflevector(blo, bhi,
                    0, 1, 2, 3, 4, 5, 6, 7, 8, 9, 10, 11, 12, 13, 14, 15);
    }
#pragma unroll
    for (int ct = 0; ct < 4; ++ct) {
      acc[0][ct] = __builtin_amdgcn_wmma_f32_16x16x32_bf16(
          false, afrag[0], false, bfrag[ct], (short)0, acc[0][ct], false, false);
      acc[1][ct] = __builtin_amdgcn_wmma_f32_16x16x32_bf16(
          false, afrag[1], false, bfrag[ct], (short)0, acc[1][ct], false, false);
    }

    if (more) store_x(nb);             // cvt + ds_store after WMMAs issued
    wait_async0();                     // next Mt tile resident in LDS
    __syncthreads();                   // single barrier per K-step
    buf = nb;
  }

  // ---- epilogue: C/D layout row = v + 8*(lane>>4), col = lane&15 ----
#pragma unroll
  for (int rt = 0; rt < 2; ++rt)
#pragma unroll
    for (int ct = 0; ct < 4; ++ct)
#pragma unroll
      for (int v = 0; v < 8; ++v) {
        const int row = b0 + rowg * 32 + rt * 16 + v + lg * 8;
        const int col = n0 + colg * 64 + ct * 16 + lh;
        out[(size_t)row * CH + col] = acc[rt][ct][v];
      }
}

// ------------------------------------------------------------------
extern "C" void kernel_launch(void* const* d_in, const int* in_sizes, int n_in,
                              void* d_out, int out_size, void* d_ws, size_t ws_size,
                              hipStream_t stream) {
  (void)in_sizes; (void)n_in; (void)out_size; (void)ws_size;
  const float* x = (const float*)d_in[0];   // (65536, 1024) f32
  const float* w = (const float*)d_in[1];   // (128, 128, 2) f32
  float* out = (float*)d_out;               // (65536, 1024) f32

  // workspace layout: P (128x1024 f32) | Q (128x1024 f32) | Mt (1024x1024 bf16) = 3 MB
  float* P = (float*)d_ws;
  float* Q = P + MODES * 1024;
  bf16_t* Mt = (bf16_t*)(Q + MODES * 1024);

  build_pq<<<dim3(128, 4), 256, 0, stream>>>(w, P, Q);
  build_m<<<dim3(1024, 4), 256, 0, stream>>>(P, Q, Mt);
  spectral_gemm<<<dim3(CH / BN, BATCH / BM), 256, 0, stream>>>(x, Mt, out);
}
